// ForgettingAttention_652835029232
// MI455X (gfx1250) — compile-verified, matrix operand order checked by reading, not run on hardware
//
#include <hip/hip_runtime.h>
#include <hip/hip_bf16.h>

// ---------------------------------------------------------------------------
// Problem constants (from reference): B=2, S=2048, D=2048, H=32, HD=64
// ---------------------------------------------------------------------------
#define BB 2
#define SS 2048
#define DD 2048
#define HH 32
#define HD 64
#define MM (BB * SS)   // 4096 rows of the "token" matrices

typedef unsigned short u16;
typedef __attribute__((ext_vector_type(16))) __bf16 v16bf;
typedef __attribute__((ext_vector_type(8)))  float  v8f;
typedef __attribute__((ext_vector_type(4)))  unsigned int tdm_g0_t;
typedef __attribute__((ext_vector_type(8)))  int          tdm_g1_t;
typedef __attribute__((ext_vector_type(4)))  int          tdm_g2_t;
typedef __attribute__((ext_vector_type(8)))  int          tdm_g4_t;

union AFrag {            // one WMMA 16x32 bf16 operand: 8 VGPRs / 32 bytes
    v16bf v;
    uint4 u[2];
};

__device__ __forceinline__ u16 f2bf(float x) {
    unsigned int u = __float_as_uint(x);
    unsigned int r = (u + 0x7FFFu + ((u >> 16) & 1u)) >> 16;   // RNE
    return (u16)r;
}

__device__ __forceinline__ v8f wmma_bf16(const v16bf& a, const v16bf& b, const v8f& c) {
    return __builtin_amdgcn_wmma_f32_16x16x32_bf16(
        /*neg_a=*/false, a, /*neg_b=*/false, b,
        /*c_mod=*/(short)0, c, /*reuse_a=*/false, /*reuse_b=*/false);
}

// low 32 bits of a generic pointer to LDS == workgroup-relative LDS offset
__device__ __forceinline__ unsigned lds_off_of(const void* p) {
    return (unsigned)(unsigned long long)p;
}

// ---------------------------------------------------------------------------
// TDM: async DMA of a 2D bf16 tile [tile_d1 rows x tile_d0 elems] from a
// row-major tensor (leading-dim d0_stride elements) into LDS (packed
// row-major). Issued once per wave; tracked with TENSORcnt.
// D# layout per CDNA5 ISA ch.8 (group0: count/lds/global/type, group1:
// data_size, tensor dims, tile dims, stride). This toolchain exposes the
// 6-arg builtin: (g0, g1, g2, g3, g4, cpol).
// ---------------------------------------------------------------------------
__device__ __forceinline__ void tdm_load_2d_bf16(
        unsigned lds_off, const u16* gptr,
        unsigned tile_d0, unsigned tile_d1,
        unsigned tensor_d0, unsigned tensor_d1, unsigned d0_stride) {
    unsigned long long ga = (unsigned long long)gptr;
    tdm_g0_t g0;
    g0[0] = 1u;                                        // count=1, user D#
    g0[1] = lds_off;                                   // lds_addr (bytes)
    g0[2] = (unsigned)(ga & 0xFFFFFFFFu);              // global_addr[31:0]
    g0[3] = (unsigned)((ga >> 32) & 0x01FFFFFFu)       // global_addr[56:32]
          | 0x80000000u;                               // type=2 ("image")
    tdm_g1_t g1;
    g1[0] = (int)(1u << 16);                           // wg_mask=0, data_size=1 (2B)
    g1[1] = (int)((tensor_d0 & 0xFFFFu) << 16);        // tensor_dim0[15:0]
    g1[2] = (int)((tensor_d0 >> 16) | ((tensor_d1 & 0xFFFFu) << 16));
    g1[3] = (int)((tensor_d1 >> 16) | (tile_d0 << 16));// tile_dim0
    g1[4] = (int)(tile_d1 & 0xFFFFu);                  // tile_dim1 (tile_dim2=0)
    g1[5] = (int)d0_stride;                            // tensor_dim0_stride[31:0]
    g1[6] = 0;                                         // stride[47:32]=0, dim1_stride lo=0
    g1[7] = 0;
    tdm_g2_t z4 = {0, 0, 0, 0};
    tdm_g4_t z8 = {0, 0, 0, 0, 0, 0, 0, 0};
    __builtin_amdgcn_tensor_load_to_lds(g0, g1, z4, z4, z8, 0);
}

// ---------------------------------------------------------------------------
// fp32 -> bf16 elementwise conversion
// ---------------------------------------------------------------------------
__global__ void cvt_f32_bf16_kernel(const float* __restrict__ in,
                                    u16* __restrict__ out, int n) {
    int i = blockIdx.x * blockDim.x + threadIdx.x;
    if (i < n) out[i] = f2bf(in[i]);
}

// ---------------------------------------------------------------------------
// f = logsigmoid(x @ Wf.T + bf) stored as [B*H, S]
// ---------------------------------------------------------------------------
__global__ void fgate_kernel(const float* __restrict__ x,
                             const float* __restrict__ Wf,
                             const float* __restrict__ bfv,
                             float* __restrict__ fout) {
    int i = blockIdx.x * blockDim.x + threadIdx.x;       // over B*S*H
    if (i >= MM * HH) return;
    int h  = i & (HH - 1);
    int ms = i >> 5;            // b*S + s
    int b  = ms / SS;
    int s  = ms - b * SS;

    const float4* xr = (const float4*)(x + (size_t)ms * DD);
    const float4* wr = (const float4*)(Wf + (size_t)h * DD);
    float acc = 0.f;
#pragma unroll 4
    for (int k = 0; k < DD / 4; ++k) {
        float4 a = xr[k], w = wr[k];
        acc += a.x * w.x + a.y * w.y + a.z * w.z + a.w * w.w;
    }
    float z = acc + bfv[h];
    // log_sigmoid(z) = min(z,0) - log1p(exp(-|z|))
    float ls = fminf(z, 0.f) - log1pf(__expf(-fabsf(z)));
    fout[((size_t)(b * HH + h)) * SS + s] = ls;
}

// ---------------------------------------------------------------------------
// c = cumsum(f, axis=S) per (b,h): one wave per row, sequential scan
// ---------------------------------------------------------------------------
__global__ void cumsum_kernel(const float* __restrict__ f, float* __restrict__ c) {
    int bh   = blockIdx.x;
    int lane = threadIdx.x;      // 0..31
    float carry = 0.f;
    for (int s0 = 0; s0 < SS; s0 += 32) {
        float v = f[(size_t)bh * SS + s0 + lane];
#pragma unroll
        for (int off = 1; off < 32; off <<= 1) {
            float t = __shfl_up(v, off, 32);
            if (lane >= off) v += t;
        }
        v += carry;
        c[(size_t)bh * SS + s0 + lane] = v;
        carry = __shfl(v, 31, 32);
    }
}

// ---------------------------------------------------------------------------
// WMMA GEMM with TDM double-buffered LDS staging.
//   C[M,N] = A[M,K] @ W[N,K]^T   (A,W bf16 row-major, K contiguous)
// Block tile 128x128, 8 waves (2 in M x 4 in N), wave tile 64x32, K-step 32.
// Per K-step, wave 0 DMAs the next 128x32 A tile and 128x32 W tile into LDS
// via TENSOR_LOAD_TO_LDS while all 8 waves run WMMAs on the current buffers.
// mode 0: C fp32 row-major [M,N]
// mode 1: C bf16 scattered to [B,H,S,HD]   (Q, K for attention)
// mode 2: C bf16 scattered to [B,H,HD,S]   (V transposed for attention)
// ---------------------------------------------------------------------------
__global__ __launch_bounds__(256) void gemm_bf16_kernel(
        const u16* __restrict__ A, const u16* __restrict__ W,
        void* __restrict__ C, int mode) {
    const int lane = threadIdx.x & 31;
    const int wave = threadIdx.x >> 5;
    const int llo  = lane & 15;
    const int lhi  = lane >> 4;        // 0/1 : K-half selector
    const int wm   = wave & 1;         // 2 waves along M
    const int wn   = wave >> 1;        // 4 waves along N
    const int mB   = blockIdx.x * 128; // block tile origin in M
    const int nB   = blockIdx.y * 128; // block tile origin in N
    const int koff = lhi * 8;

    __shared__ __align__(16) u16 ldsA[2][128 * 32];   // 2 x 8KB
    __shared__ __align__(16) u16 ldsB[2][128 * 32];   // 2 x 8KB

    v8f acc[4][2];
#pragma unroll
    for (int mi = 0; mi < 4; ++mi)
#pragma unroll
        for (int ni = 0; ni < 2; ++ni)
#pragma unroll
            for (int r = 0; r < 8; ++r) acc[mi][ni][r] = 0.f;

    // prologue: DMA first K-slab into buffer 0
    if (wave == 0) {
        tdm_load_2d_bf16(lds_off_of(&ldsA[0][0]), A + (size_t)mB * DD,
                         32, 128, DD, MM, DD);
        tdm_load_2d_bf16(lds_off_of(&ldsB[0][0]), W + (size_t)nB * DD,
                         32, 128, DD, DD, DD);
    }

    const int nIter = DD / 32;
    for (int it = 0; it < nIter; ++it) {
        const int cur = it & 1;
        if (wave == 0) __builtin_amdgcn_s_wait_tensorcnt(0);
        __syncthreads();                       // buffers[cur] ready for all waves
        if (it + 1 < nIter && wave == 0) {     // DMA next slab while we compute
            const int kn = (it + 1) * 32;
            tdm_load_2d_bf16(lds_off_of(&ldsA[cur ^ 1][0]),
                             A + (size_t)mB * DD + kn, 32, 128, DD, MM, DD);
            tdm_load_2d_bf16(lds_off_of(&ldsB[cur ^ 1][0]),
                             W + (size_t)nB * DD + kn, 32, 128, DD, DD, DD);
        }

        AFrag a[4], b[2];
#pragma unroll
        for (int mi = 0; mi < 4; ++mi) {
            const u16* ar = &ldsA[cur][(wm * 64 + mi * 16 + llo) * 32];
            a[mi].u[0] = *(const uint4*)(ar + koff);
            a[mi].u[1] = *(const uint4*)(ar + 16 + koff);
        }
#pragma unroll
        for (int ni = 0; ni < 2; ++ni) {
            const u16* br = &ldsB[cur][(wn * 32 + ni * 16 + llo) * 32];
            b[ni].u[0] = *(const uint4*)(br + koff);
            b[ni].u[1] = *(const uint4*)(br + 16 + koff);
        }
#pragma unroll
        for (int mi = 0; mi < 4; ++mi)
#pragma unroll
            for (int ni = 0; ni < 2; ++ni)
                acc[mi][ni] = wmma_bf16(a[mi].v, b[ni].v, acc[mi][ni]);
        __syncthreads();                       // done reading buffers[cur]
    }

    // epilogue
#pragma unroll
    for (int mi = 0; mi < 4; ++mi) {
#pragma unroll
        for (int ni = 0; ni < 2; ++ni) {
#pragma unroll
            for (int r = 0; r < 8; ++r) {
                int row = mB + wm * 64 + mi * 16 + r + 8 * lhi;  // token b*S+s
                int col = nB + wn * 32 + ni * 16 + llo;          // feature
                float val = acc[mi][ni][r];
                if (mode == 0) {
                    ((float*)C)[(size_t)row * DD + col] = val;
                } else {
                    int b = row / SS, s = row - b * SS;
                    int h = col >> 6, d = col & 63;
                    u16* out = (u16*)C;
                    if (mode == 1)       // [B,H,S,HD]
                        out[(((size_t)(b * HH + h)) * SS + s) * HD + d] = f2bf(val);
                    else                 // [B,H,HD,S]
                        out[(((size_t)(b * HH + h)) * HD + d) * SS + s] = f2bf(val);
                }
            }
        }
    }
}

// ---------------------------------------------------------------------------
// Flash attention with forget gates.
//   Q,K : bf16 [B*H, S, 64]    Vt : bf16 [B*H, 64, S]    c : f32 [B*H, S]
//   O   : bf16 [B*S, D]  (row = b*S+s, col = h*64+d) for the final projection
// Each wave owns 16 query rows; key tiles of 32; online softmax.
// ---------------------------------------------------------------------------
__global__ __launch_bounds__(256) void fattn_kernel(
        const u16* __restrict__ Q, const u16* __restrict__ Kk,
        const u16* __restrict__ Vt, const float* __restrict__ c,
        u16* __restrict__ O) {
    const int bh   = blockIdx.y;
    const int b    = bh / HH;
    const int h    = bh - b * HH;
    const int lane = threadIdx.x & 31;
    const int wave = threadIdx.x >> 5;
    const int llo  = lane & 15;
    const int lhi  = lane >> 4;
    const int q0   = blockIdx.x * 128 + wave * 16;   // wave's 16 query rows
    const float scale = 0.125f;                      // 1/sqrt(64)

    const u16*   Qb  = Q  + (size_t)bh * SS * HD;
    const u16*   Kb  = Kk + (size_t)bh * SS * HD;
    const u16*   Vb  = Vt + (size_t)bh * HD * SS;
    const float* cb  = c  + (size_t)bh * SS;

    __shared__ __align__(16) u16 pshm[8][16 * 32];   // per-wave P scratch (1KB)

    // Q fragments for the two 32-wide d-slices of HD=64
    AFrag qf[2];
#pragma unroll
    for (int di = 0; di < 2; ++di) {
        const u16* qr = Qb + (size_t)(q0 + llo) * HD + di * 32 + lhi * 8;
        qf[di].u[0] = *(const uint4*)(qr);
        qf[di].u[1] = *(const uint4*)(qr + 16);
    }

    // per-row forget-gate prefix sums for this wave's rows
    float ci[8];
#pragma unroll
    for (int r = 0; r < 8; ++r) ci[r] = cb[q0 + r + 8 * lhi];

    float m_i[8], l_i[8];
#pragma unroll
    for (int r = 0; r < 8; ++r) { m_i[r] = -1e30f; l_i[r] = 0.f; }
    v8f accO[4];
#pragma unroll
    for (int n = 0; n < 4; ++n)
#pragma unroll
        for (int r = 0; r < 8; ++r) accO[n][r] = 0.f;

    for (int j0 = 0; j0 <= q0 + 15; j0 += 32) {
        // ---- scores: S = Q K^T over HD=64 (two k-steps), two 16-key N tiles
        v8f sc[2];
#pragma unroll
        for (int jn = 0; jn < 2; ++jn)
#pragma unroll
            for (int r = 0; r < 8; ++r) sc[jn][r] = 0.f;
#pragma unroll
        for (int jn = 0; jn < 2; ++jn) {
#pragma unroll
            for (int di = 0; di < 2; ++di) {
                AFrag kf;
                const u16* kr = Kb + (size_t)(j0 + jn * 16 + llo) * HD + di * 32 + lhi * 8;
                kf.u[0] = *(const uint4*)(kr);
                kf.u[1] = *(const uint4*)(kr + 16);
                sc[jn] = wmma_bf16(qf[di].v, kf.v, sc[jn]);
            }
        }

        // ---- forget bias + causal mask
        float cj[2];
#pragma unroll
        for (int jn = 0; jn < 2; ++jn) cj[jn] = cb[j0 + jn * 16 + llo];

        float sv[2][8];
#pragma unroll
        for (int jn = 0; jn < 2; ++jn) {
            int jcol = j0 + jn * 16 + llo;
#pragma unroll
            for (int r = 0; r < 8; ++r) {
                int qrow = q0 + r + 8 * lhi;
                float v = sc[jn][r] * scale + ci[r] - cj[jn];
                sv[jn][r] = (jcol <= qrow) ? v : -1e30f;
            }
        }

        // ---- online softmax: row-reduce across the 16 lanes of each half
        float tm[8];
#pragma unroll
        for (int r = 0; r < 8; ++r) tm[r] = fmaxf(sv[0][r], sv[1][r]);
#pragma unroll
        for (int off = 1; off < 16; off <<= 1)
#pragma unroll
            for (int r = 0; r < 8; ++r)
                tm[r] = fmaxf(tm[r], __shfl_xor(tm[r], off, 32));

        float alpha[8], rs[8];
#pragma unroll
        for (int r = 0; r < 8; ++r) {
            float mn = fmaxf(m_i[r], tm[r]);
            alpha[r] = __expf(m_i[r] - mn);
            m_i[r] = mn;
        }
        float pv[2][8];
#pragma unroll
        for (int jn = 0; jn < 2; ++jn)
#pragma unroll
            for (int r = 0; r < 8; ++r)
                pv[jn][r] = __expf(sv[jn][r] - m_i[r]);
#pragma unroll
        for (int r = 0; r < 8; ++r) rs[r] = pv[0][r] + pv[1][r];
#pragma unroll
        for (int off = 1; off < 16; off <<= 1)
#pragma unroll
            for (int r = 0; r < 8; ++r)
                rs[r] += __shfl_xor(rs[r], off, 32);
#pragma unroll
        for (int r = 0; r < 8; ++r) l_i[r] = l_i[r] * alpha[r] + rs[r];
#pragma unroll
        for (int n = 0; n < 4; ++n)
#pragma unroll
            for (int r = 0; r < 8; ++r) accO[n][r] *= alpha[r];

        // ---- D-layout -> A-layout for P via per-wave LDS scratch (bf16)
#pragma unroll
        for (int jn = 0; jn < 2; ++jn)
#pragma unroll
            for (int r = 0; r < 8; ++r)
                pshm[wave][(r + 8 * lhi) * 32 + jn * 16 + llo] = f2bf(pv[jn][r]);

        AFrag pf;
        pf.u[0] = *(const uint4*)&pshm[wave][llo * 32 + lhi * 8];
        pf.u[1] = *(const uint4*)&pshm[wave][llo * 32 + 16 + lhi * 8];

        // ---- O += P @ V  (Vt rows are d, contiguous along keys)
#pragma unroll
        for (int n = 0; n < 4; ++n) {
            AFrag vf;
            const u16* vr = Vb + (size_t)(n * 16 + llo) * SS + j0 + lhi * 8;
            vf.u[0] = *(const uint4*)(vr);
            vf.u[1] = *(const uint4*)(vr + 16);
            accO[n] = wmma_bf16(pf.v, vf.v, accO[n]);
        }
    }

    // ---- normalize and store bf16 [B*S, D]
    float inv[8];
#pragma unroll
    for (int r = 0; r < 8; ++r) inv[r] = 1.0f / l_i[r];
#pragma unroll
    for (int n = 0; n < 4; ++n)
#pragma unroll
        for (int r = 0; r < 8; ++r) {
            int srow = q0 + r + 8 * lhi;
            int col  = h * HD + n * 16 + llo;
            O[((size_t)(b * SS + srow)) * DD + col] = f2bf(accO[n][r] * inv[r]);
        }
}

// ---------------------------------------------------------------------------
// Host-side launch
// ---------------------------------------------------------------------------
extern "C" void kernel_launch(void* const* d_in, const int* in_sizes, int n_in,
                              void* d_out, int out_size, void* d_ws, size_t ws_size,
                              hipStream_t stream) {
    const float* x  = (const float*)d_in[0];
    const float* Wq = (const float*)d_in[1];
    const float* Wk = (const float*)d_in[2];
    const float* Wv = (const float*)d_in[3];
    const float* Wf = (const float*)d_in[4];
    const float* bf = (const float*)d_in[5];
    const float* Wo = (const float*)d_in[6];

    // workspace carve-up (~117 MB total)
    size_t off = 0;
    auto carve = [&](size_t bytes) {
        void* p = (char*)d_ws + off;
        off += (bytes + 255) & ~(size_t)255;
        return p;
    };
    u16*   xbf  = (u16*)carve((size_t)MM * DD * 2);
    u16*   wqb  = (u16*)carve((size_t)DD * DD * 2);
    u16*   wkb  = (u16*)carve((size_t)DD * DD * 2);
    u16*   wvb  = (u16*)carve((size_t)DD * DD * 2);
    u16*   wob  = (u16*)carve((size_t)DD * DD * 2);
    u16*   Qbf  = (u16*)carve((size_t)MM * DD * 2);
    u16*   Kbf  = (u16*)carve((size_t)MM * DD * 2);
    u16*   Vtb  = (u16*)carve((size_t)MM * DD * 2);
    float* fbuf = (float*)carve((size_t)BB * HH * SS * 4);
    float* cbuf = (float*)carve((size_t)BB * HH * SS * 4);
    u16*   Obf  = (u16*)carve((size_t)MM * DD * 2);
    (void)ws_size;

    const int nx = MM * DD, nw = DD * DD;
    cvt_f32_bf16_kernel<<<(nx + 255) / 256, 256, 0, stream>>>(x, xbf, nx);
    cvt_f32_bf16_kernel<<<(nw + 255) / 256, 256, 0, stream>>>(Wq, wqb, nw);
    cvt_f32_bf16_kernel<<<(nw + 255) / 256, 256, 0, stream>>>(Wk, wkb, nw);
    cvt_f32_bf16_kernel<<<(nw + 255) / 256, 256, 0, stream>>>(Wv, wvb, nw);
    cvt_f32_bf16_kernel<<<(nw + 255) / 256, 256, 0, stream>>>(Wo, wob, nw);

    fgate_kernel<<<(MM * HH + 255) / 256, 256, 0, stream>>>(x, Wf, bf, fbuf);
    cumsum_kernel<<<BB * HH, 32, 0, stream>>>(fbuf, cbuf);

    dim3 ggrid(MM / 128, DD / 128);
    gemm_bf16_kernel<<<ggrid, 256, 0, stream>>>(xbf, wqb, Qbf, 1);
    gemm_bf16_kernel<<<ggrid, 256, 0, stream>>>(xbf, wkb, Kbf, 1);
    gemm_bf16_kernel<<<ggrid, 256, 0, stream>>>(xbf, wvb, Vtb, 2);

    dim3 agrid(SS / 128, BB * HH);
    fattn_kernel<<<agrid, 256, 0, stream>>>(Qbf, Kbf, Vtb, cbuf, Obf);

    gemm_bf16_kernel<<<ggrid, 256, 0, stream>>>(Obf, wob, d_out, 0);
}
